// Attention_39479339385018
// MI455X (gfx1250) — matrix operand order repaired and checked by reading, not compiled, not run on hardware
//
#include <hip/hip_runtime.h>

// ---------------------------------------------------------------------------
// MI455X (gfx1250) attention block: bf16 WMMA GEMMs + flash attention.
// wave32, 256-thread (8-wave) workgroups. Async global->LDS double buffering.
// ---------------------------------------------------------------------------

typedef __attribute__((ext_vector_type(16))) __bf16 v16bf;
typedef __attribute__((ext_vector_type(8)))  __bf16 v8bf;
typedef __attribute__((ext_vector_type(8)))  float  v8f;
typedef __attribute__((ext_vector_type(4)))  int    v4i;

constexpr int DMODEL = 4096;
constexpr int NHEADS = 32;
constexpr int HDIM   = 128;
constexpr int BATCH  = 2;
constexpr int SEQ    = 2048;
constexpr int BS     = BATCH * SEQ;   // 4096 activation rows

// ---------------------------------------------------------------------------
// CDNA5 async global->LDS copy (ASYNCcnt) with synchronous fallback
// ---------------------------------------------------------------------------
#if defined(__HIP_DEVICE_COMPILE__) && __has_builtin(__builtin_amdgcn_global_load_async_to_lds_b128)
#define HAS_ASYNC_LDS 1
#else
#define HAS_ASYNC_LDS 0
#endif

#define AS_GLOBAL __attribute__((address_space(1)))
#define AS_LDS    __attribute__((address_space(3)))

__device__ inline void async_copy16(const __bf16* g, __bf16* l) {
#if HAS_ASYNC_LDS
    // two-step casts: const drop -> addrspacecast (void) -> pointee reinterpret
    AS_GLOBAL void* gv = (AS_GLOBAL void*)const_cast<__bf16*>(g);
    AS_LDS    void* lv = (AS_LDS    void*)l;
    __builtin_amdgcn_global_load_async_to_lds_b128(
        (AS_GLOBAL v4i*)gv, (AS_LDS v4i*)lv, 0, 0);
#else
    *(v8bf*)l = *(const v8bf*)g;
#endif
}

__device__ inline void async_wait_all() {
#if HAS_ASYNC_LDS
#if __has_builtin(__builtin_amdgcn_s_wait_asynccnt)
    __builtin_amdgcn_s_wait_asynccnt(0);
#else
    asm volatile("s_wait_asynccnt 0" ::: "memory");
#endif
#endif
}

// ---------------------------------------------------------------------------
// WMMA helpers
// ---------------------------------------------------------------------------
__device__ inline v8f wmma_bf16(v16bf a, v16bf b, v8f c) {
    return __builtin_amdgcn_wmma_f32_16x16x32_bf16(
        false, a, false, b, (short)0, c, false, false);
}

__device__ inline v16bf frag2(const __bf16* p0, const __bf16* p1) {
    v8bf lo = *(const v8bf*)p0;
    v8bf hi = *(const v8bf*)p1;
    v16bf f;
#pragma unroll
    for (int i = 0; i < 8; ++i) { f[i] = lo[i]; f[8 + i] = hi[i]; }
    return f;
}

// ---------------------------------------------------------------------------
// fp32 -> bf16 conversion
// ---------------------------------------------------------------------------
__global__ __launch_bounds__(256) void cvt_f32_bf16(const float* __restrict__ in,
                                                    __bf16* __restrict__ out, int n) {
    int i = (blockIdx.x * blockDim.x + threadIdx.x) * 4;
    if (i + 3 < n) {
        float4 v = *(const float4*)(in + i);
        out[i + 0] = (__bf16)v.x;
        out[i + 1] = (__bf16)v.y;
        out[i + 2] = (__bf16)v.z;
        out[i + 3] = (__bf16)v.w;
    }
}

// ---------------------------------------------------------------------------
// NT GEMM (4096x4096x4096): C[m,n] = sum_k A[m,k] * W[n,k], fp32 out.
// Block tile 128x128, BK=32, double-buffered LDS fed by async copies.
// Each of 8 waves computes a 64x32 patch (4x2 WMMA tiles).
// ---------------------------------------------------------------------------
__global__ __launch_bounds__(256) void gemm_nt_bf16(const __bf16* __restrict__ A,
                                                    const __bf16* __restrict__ W,
                                                    float* __restrict__ C) {
    constexpr int LDT = 32 + 8;                   // padded row stride (16B aligned)
    __shared__ __bf16 As[2][128][LDT];
    __shared__ __bf16 Bs[2][128][LDT];

    const int tid  = threadIdx.x;
    const int wave = tid >> 5;
    const int lane = tid & 31;
    const int half = lane >> 4;
    const int r    = lane & 15;
    const int waveM = wave & 1;
    const int waveN = wave >> 1;
    const int bm = blockIdx.y;
    const int bn = blockIdx.x;

    // per-thread cooperative-load coordinates (2 chunks of 16B per tile)
    int rowc[2], offc[2];
#pragma unroll
    for (int c = 0; c < 2; ++c) {
        int ci = tid + 256 * c;                   // 512 chunks per 128x32 tile
        rowc[c] = ci >> 2;
        offc[c] = (ci & 3) * 8;
    }

    auto issue_tile = [&](int kb, int buf) {
#pragma unroll
        for (int c = 0; c < 2; ++c) {
            const __bf16* ga = A + (bm * 128 + rowc[c]) * DMODEL + kb * 32 + offc[c];
            const __bf16* gw = W + (bn * 128 + rowc[c]) * DMODEL + kb * 32 + offc[c];
            async_copy16(ga, &As[buf][rowc[c]][offc[c]]);
            async_copy16(gw, &Bs[buf][rowc[c]][offc[c]]);
        }
    };

    v8f acc[4][2];
#pragma unroll
    for (int mt = 0; mt < 4; ++mt)
#pragma unroll
        for (int nt = 0; nt < 2; ++nt) acc[mt][nt] = (v8f)0.0f;

    constexpr int ksteps = DMODEL / 32;           // 128
    issue_tile(0, 0);

    for (int kb = 0; kb < ksteps; ++kb) {
        const int buf = kb & 1;
        async_wait_all();                         // my chunks of tile kb landed
        __syncthreads();                          // everyone's landed; prev buf free
        if (kb + 1 < ksteps) {
            issue_tile(kb + 1, buf ^ 1);          // overlap next copy with compute
            // L2 prefetch a few tiles ahead
            __builtin_prefetch(A + (bm * 128 + rowc[0]) * DMODEL + (kb + 4) * 32, 0, 0);
            __builtin_prefetch(W + (bn * 128 + rowc[0]) * DMODEL + (kb + 4) * 32, 0, 0);
        }

        v16bf af[4], bfr[2];
#pragma unroll
        for (int mt = 0; mt < 4; ++mt) {
            const __bf16* p = &As[buf][waveM * 64 + mt * 16 + r][0];
            af[mt] = frag2(p + half * 8, p + half * 8 + 16);   // A: interleaved octets
        }
#pragma unroll
        for (int nt = 0; nt < 2; ++nt) {
            const __bf16* p = &Bs[buf][waveN * 32 + nt * 16 + r][0];
            bfr[nt] = frag2(p + half * 8, p + half * 8 + 16);  // B: interleaved octets
        }
#pragma unroll
        for (int mt = 0; mt < 4; ++mt)
#pragma unroll
            for (int nt = 0; nt < 2; ++nt)
                acc[mt][nt] = wmma_bf16(af[mt], bfr[nt], acc[mt][nt]);
    }

    // epilogue (32-bit index math; C row = j + 8*half, col = r)
    const int colBase = bn * 128 + waveN * 32 + r;
#pragma unroll
    for (int mt = 0; mt < 4; ++mt) {
        const int rowBase = bm * 128 + waveM * 64 + mt * 16 + 8 * half;
#pragma unroll
        for (int j = 0; j < 8; ++j) {
            float* crow = C + (rowBase + j) * DMODEL + colBase;
#pragma unroll
            for (int nt = 0; nt < 2; ++nt) crow[nt * 16] = acc[mt][nt][j];
        }
    }
}

// ---------------------------------------------------------------------------
// RoPE on Q,K (fp32 in) -> bf16 out; V fp32 -> bf16 passthrough.
// ---------------------------------------------------------------------------
__global__ __launch_bounds__(256) void rope_pack(const float* __restrict__ Qf,
                                                 const float* __restrict__ Kf,
                                                 const float* __restrict__ Vf,
                                                 const float* __restrict__ cosT,
                                                 const float* __restrict__ sinT,
                                                 __bf16* __restrict__ Qb,
                                                 __bf16* __restrict__ Kb,
                                                 __bf16* __restrict__ Vb) {
    const int HP = HDIM / 2;                      // 64 pairs
    int tid = blockIdx.x * blockDim.x + threadIdx.x;
    if (tid >= BS * NHEADS * HP) return;
    int i  = tid % HP;
    int h  = (tid / HP) % NHEADS;
    int bs = tid / (HP * NHEADS);
    int s  = bs % SEQ;
    int base = (bs * NHEADS + h) * HDIM + 2 * i;  // < 2^25, int-safe
    float c  = cosT[s * HP + i];
    float sn = sinT[s * HP + i];

    float qe = Qf[base], qo = Qf[base + 1];
    Qb[base]     = (__bf16)(qe * c - qo * sn);
    Qb[base + 1] = (__bf16)(qe * sn + qo * c);

    float ke = Kf[base], ko = Kf[base + 1];
    Kb[base]     = (__bf16)(ke * c - ko * sn);
    Kb[base + 1] = (__bf16)(ke * sn + ko * c);

    Vb[base]     = (__bf16)Vf[base];
    Vb[base + 1] = (__bf16)Vf[base + 1];
}

// ---------------------------------------------------------------------------
// Flash attention: block = 128 queries x one (b,h). 8 waves x 16 queries.
// Online softmax over 64-key tiles; K tile staged via async copy, V tile
// transposed through VALU (overlaps the async engine).
// ---------------------------------------------------------------------------
__global__ __launch_bounds__(256) void flash_attn(const __bf16* __restrict__ Q,
                                                  const __bf16* __restrict__ K,
                                                  const __bf16* __restrict__ V,
                                                  __bf16* __restrict__ O) {
    constexpr int KT  = 64;
    constexpr int LDK = HDIM + 8;                 // 136
    constexpr int LDV = KT + 8;                   // 72
    __shared__ __bf16 Kt[KT][LDK];
    __shared__ __bf16 Vt[HDIM][LDV];
    __shared__ __bf16 Pt[8][16][LDV];

    const int tid  = threadIdx.x;
    const int wave = tid >> 5;
    const int lane = tid & 31;
    const int half = lane >> 4;
    const int r    = lane & 15;
    const int qblk = blockIdx.x;
    const int bh   = blockIdx.y;
    const int b    = bh >> 5;
    const int h    = bh & 31;
    const int qrow0 = qblk * 128 + wave * 16;

    v16bf qf[4];
    const __bf16* qbase = Q + ((b * SEQ + qrow0 + r) * NHEADS + h) * HDIM;
#pragma unroll
    for (int kc = 0; kc < 4; ++kc)
        qf[kc] = frag2(qbase + kc * 32 + half * 8, qbase + kc * 32 + half * 8 + 16);

    v8f o[8];
#pragma unroll
    for (int nd = 0; nd < 8; ++nd) o[nd] = (v8f)0.0f;
    float m[8], l[8];
#pragma unroll
    for (int j = 0; j < 8; ++j) { m[j] = -INFINITY; l[j] = 0.0f; }
    const float scale = 0.08838834764831845f;     // 1/sqrt(128)

    for (int kt = 0; kt < SEQ / KT; ++kt) {
        __syncthreads();
#pragma unroll
        for (int c = 0; c < 4; ++c) {
            int ci  = tid + 256 * c;              // 1024 chunks of 8 bf16
            int row = ci >> 4;
            int off = (ci & 15) * 8;
            int g   = ((b * SEQ + kt * KT + row) * NHEADS + h) * HDIM + off;
            async_copy16(K + g, &Kt[row][off]);   // straight copy: async engine
            v8bf vv = *(const v8bf*)(V + g);      // transpose path: VALU
#pragma unroll
            for (int j = 0; j < 8; ++j) Vt[off + j][row] = vv[j];
        }
        async_wait_all();
        __syncthreads();

        // S = Q * K^T
        v8f sc[4];
#pragma unroll
        for (int nt = 0; nt < 4; ++nt) sc[nt] = (v8f)0.0f;
#pragma unroll
        for (int nt = 0; nt < 4; ++nt) {
            const __bf16* p = &Kt[nt * 16 + r][0];
#pragma unroll
            for (int kc = 0; kc < 4; ++kc) {
                v16bf kf = frag2(p + kc * 32 + half * 8, p + kc * 32 + half * 8 + 16);
                sc[nt] = wmma_bf16(qf[kc], kf, sc[nt]);
            }
        }
#pragma unroll
        for (int nt = 0; nt < 4; ++nt)
#pragma unroll
            for (int j = 0; j < 8; ++j) sc[nt][j] *= scale;

        // online softmax stats (rows j + 8*half, 16-lane-group reductions)
        float alpha[8];
#pragma unroll
        for (int j = 0; j < 8; ++j) {
            float v = fmaxf(fmaxf(sc[0][j], sc[1][j]), fmaxf(sc[2][j], sc[3][j]));
#pragma unroll
            for (int d = 1; d < 16; d <<= 1) v = fmaxf(v, __shfl_xor(v, d, 32));
            float mn = fmaxf(m[j], v);
            alpha[j] = __expf(m[j] - mn);
            m[j] = mn;
            float rs = 0.0f;
#pragma unroll
            for (int nt = 0; nt < 4; ++nt) {
                float p = __expf(sc[nt][j] - mn);
                sc[nt][j] = p;
                rs += p;
            }
#pragma unroll
            for (int d = 1; d < 16; d <<= 1) rs += __shfl_xor(rs, d, 32);
            l[j] = l[j] * alpha[j] + rs;
        }
#pragma unroll
        for (int nd = 0; nd < 8; ++nd)
#pragma unroll
            for (int j = 0; j < 8; ++j) o[nd][j] *= alpha[j];

        // P (C layout) -> per-wave LDS -> A-fragment layout
#pragma unroll
        for (int nt = 0; nt < 4; ++nt)
#pragma unroll
            for (int j = 0; j < 8; ++j)
                Pt[wave][j + 8 * half][nt * 16 + r] = (__bf16)sc[nt][j];

        // O += P * V
#pragma unroll
        for (int kc = 0; kc < 2; ++kc) {
            const __bf16* pp = &Pt[wave][r][0];
            v16bf pf = frag2(pp + kc * 32 + half * 8, pp + kc * 32 + half * 8 + 16);
#pragma unroll
            for (int nd = 0; nd < 8; ++nd) {
                const __bf16* vp = &Vt[nd * 16 + r][0];
                v16bf vf = frag2(vp + kc * 32 + half * 8, vp + kc * 32 + half * 8 + 16);
                o[nd] = wmma_bf16(pf, vf, o[nd]);
            }
        }
    }

    // normalize, store attention output (bf16, [B,S,NH*HD])
#pragma unroll
    for (int j = 0; j < 8; ++j) {
        float inv = 1.0f / l[j];
        int row = qrow0 + j + 8 * half;
#pragma unroll
        for (int nd = 0; nd < 8; ++nd)
            O[((b * SEQ + row) * NHEADS + h) * HDIM + nd * 16 + r] =
                (__bf16)(o[nd][j] * inv);
    }
}

// ---------------------------------------------------------------------------
// host: stage workspace, launch pipeline
// ---------------------------------------------------------------------------
extern "C" void kernel_launch(void* const* d_in, const int* in_sizes, int n_in,
                              void* d_out, int out_size, void* d_ws, size_t ws_size,
                              hipStream_t stream) {
    (void)in_sizes; (void)n_in; (void)out_size; (void)ws_size;

    const float* x    = (const float*)d_in[0];
    const float* cosT = (const float*)d_in[2];
    const float* sinT = (const float*)d_in[3];
    const float* wq   = (const float*)d_in[4];
    const float* wk   = (const float*)d_in[5];
    const float* wv   = (const float*)d_in[6];
    const float* wo   = (const float*)d_in[7];
    float* out = (float*)d_out;

    const size_t E = (size_t)DMODEL * DMODEL;     // 16.7M elements (== BS*DMODEL)

    char* w = (char*)d_ws;
    auto alloc = [&](size_t bytes) -> void* {
        void* p = (void*)w;
        w += (bytes + 255) & ~(size_t)255;
        return p;
    };
    __bf16* xb  = (__bf16*)alloc(E * 2);
    __bf16* wqb = (__bf16*)alloc(E * 2);
    __bf16* wkb = (__bf16*)alloc(E * 2);
    __bf16* wvb = (__bf16*)alloc(E * 2);
    __bf16* wob = (__bf16*)alloc(E * 2);
    float*  Qf  = (float*)alloc(E * 4);
    float*  Kf  = (float*)alloc(E * 4);
    float*  Vf  = (float*)alloc(E * 4);
    __bf16* Qb  = (__bf16*)alloc(E * 2);
    __bf16* Kb  = (__bf16*)alloc(E * 2);
    __bf16* Vb  = (__bf16*)alloc(E * 2);
    __bf16* Ob  = (__bf16*)alloc(E * 2);

    const int cvtBlocks = (int)(E / 1024);
    cvt_f32_bf16<<<cvtBlocks, 256, 0, stream>>>(x,  xb,  (int)E);
    cvt_f32_bf16<<<cvtBlocks, 256, 0, stream>>>(wq, wqb, (int)E);
    cvt_f32_bf16<<<cvtBlocks, 256, 0, stream>>>(wk, wkb, (int)E);
    cvt_f32_bf16<<<cvtBlocks, 256, 0, stream>>>(wv, wvb, (int)E);
    cvt_f32_bf16<<<cvtBlocks, 256, 0, stream>>>(wo, wob, (int)E);

    dim3 gGemm(DMODEL / 128, BS / 128);           // (32, 32)
    gemm_nt_bf16<<<gGemm, 256, 0, stream>>>(xb, wqb, Qf);
    gemm_nt_bf16<<<gGemm, 256, 0, stream>>>(xb, wkb, Kf);
    gemm_nt_bf16<<<gGemm, 256, 0, stream>>>(xb, wvb, Vf);

    const int ropeTotal = BS * NHEADS * (HDIM / 2);
    rope_pack<<<ropeTotal / 256, 256, 0, stream>>>(Qf, Kf, Vf, cosT, sinT, Qb, Kb, Vb);

    dim3 gFlash(SEQ / 128, BATCH * NHEADS);       // (16, 64)
    flash_attn<<<gFlash, 256, 0, stream>>>(Qb, Kb, Vb, Ob);

    gemm_nt_bf16<<<gGemm, 256, 0, stream>>>(Ob, wob, out);
}
